// Model_46273977647543
// MI455X (gfx1250) — matrix operand order repaired and checked by reading, not compiled, not run on hardware
//
#include <hip/hip_runtime.h>

// ---------------------------------------------------------------------------
// MI455X (gfx1250, wave32) fused implementation.
//   - 1 workgroup (256 threads = 8 waves) per batch element, all activations
//     resident in the 320KB WGP LDS.
//   - every GEMM on v_wmma_f32_16x16x32_bf16, f32 accumulate; the LayerNorms
//     are fused into the A-fragment gathers (VALU co-executes with WMMA).
//   - explicit addrspace(3)/(1) pointers: ds_load_b128 for LDS fragments,
//     global_load_b128 for weights, no generic-pointer null checks.
// ---------------------------------------------------------------------------

typedef __bf16 bf16_t;
typedef __attribute__((ext_vector_type(16))) __bf16 v16bf;
typedef __attribute__((ext_vector_type(8)))  __bf16 v8bf;
typedef __attribute__((ext_vector_type(8)))  float  v8f;
typedef __attribute__((ext_vector_type(4)))  float  v4f;

#define AS3 __attribute__((address_space(3)))
#define AS1 __attribute__((address_space(1)))

#define NTHREADS 256
#define NWAVES   8

__host__ __device__ static inline int pad32i(int x) { return (x + 31) & ~31; }

// ----------------------------- LDS layout (bytes) ---------------------------
#define A_BYTES   204288
#define TN_OFF    114688
#define H_OFF     A_BYTES
#define H_BYTES   89600
#define CCRAW_OFF (H_OFF + 45056)
#define P_OFF     (H_OFF + H_BYTES)
#define P_BYTES   25600
#define SM_OFF    (P_OFF + P_BYTES)
#define SM_BYTES  6528
#define LDS_TOTAL (SM_OFF + SM_BYTES)
static_assert(LDS_TOTAL <= 327680, "exceeds gfx1250 320KB LDS");

// ----------------------------- kernel parameters ----------------------------
struct KParams {
  const float* x;
  const int*   alpha;
  float*       out;
  const float* embed_b;
  // LayerNorm params: zero-padded f32 copies in ws (length pad32(C))
  const float* ccg_lnw; const float* ccg_lnb; const float* ccg_pb;
  const float* n0w[4]; const float* n0b[4];
  const float* redb[4];
  const float* n1w[4]; const float* n1b[4];
  const float* qb[4]; const float* kb[4]; const float* vb[4];
  const float* cs_lnw[4]; const float* cs_lnb[4];
  const float* thw[4]; const float* thb[4];
  const float* cs_pb[3];
  const float* normw; const float* normb; const float* headw; const float* headb;
  // bf16, transposed [out][pad32(in)] weights in ws
  const bf16_t* wt_embed; const bf16_t* wt_ccg;
  const bf16_t* wt_red[4]; const bf16_t* wt_q[4]; const bf16_t* wt_k[4]; const bf16_t* wt_v[4];
  const bf16_t* wt_cp[3];
};

// ------------------------- address-space load helpers ------------------------
__device__ __forceinline__ v8bf ld8(const AS3 bf16_t* p) { return *(const AS3 v8bf*)p; }
__device__ __forceinline__ v8bf ld8(const AS1 bf16_t* p) { return *(const AS1 v8bf*)p; }
__device__ __forceinline__ v4f  ld4f(const AS1 float* p) { return *(const AS1 v4f*)p; }

union FragU { v16bf v; v8bf h[2]; bf16_t e[16]; };

// ----------------------------- fragment loaders -----------------------------
// A 16x32 bf16 fragment (row-major src). lanes 0-15: M=lane, K=0..7 & 16..23;
// lanes 16-31: M=lane-16, K=8..15 & 24..31.
template <typename PB>
__device__ __forceinline__ v16bf frag_ld_a(PB M, int ld, int r0, int k0) {
  int l = threadIdx.x & 31;
  PB p = M + (size_t)(r0 + (l & 15)) * ld + (k0 + ((l & 16) ? 8 : 0));
  FragU u;
  u.h[0] = ld8(p);
  u.h[1] = ld8(p + 16);
  return u.v;
}

// B 32x16 bf16 fragment; src row n holds B[:,n] contiguously (src is B^T
// row-major). lanes 0-15: N=lane, K=0..7 & 16..23; lanes 16-31: K=8..15 & 24..31.
template <typename PB>
__device__ __forceinline__ v16bf frag_ld_b(PB M, int ld, int n0, int k0) {
  int l = threadIdx.x & 31;
  PB p = M + (size_t)(n0 + (l & 15)) * ld + (k0 + ((l & 16) ? 8 : 0));
  FragU u;
  u.h[0] = ld8(p);
  u.h[1] = ld8(p + 16);
  return u.v;
}

// A fragment with LayerNorm fused into the gather. Branch-free: w/b are
// zero-padded to pad32(C), M's pad columns hold finite bf16, so pads yield 0.
__device__ __forceinline__ v16bf frag_ld_a_ln(const AS3 bf16_t* M, int ld, int r0, int k0,
                                              const AS3 float* st,
                                              const AS1 float* w, const AS1 float* b) {
  int l   = threadIdx.x & 31;
  int row = r0 + (l & 15);
  int kb  = k0 + ((l & 16) ? 8 : 0);
  float mean = st[2 * row], rstd = st[2 * row + 1];
  FragU u;
#pragma unroll
  for (int hh = 0; hh < 2; ++hh) {
    int base = kb + hh * 16;
    v8bf xv = ld8(M + (size_t)row * ld + base);
    v4f w0 = ld4f(w + base), w1 = ld4f(w + base + 4);
    v4f b0 = ld4f(b + base), b1 = ld4f(b + base + 4);
#pragma unroll
    for (int j = 0; j < 4; ++j) {
      u.e[hh * 8 + j]     = (bf16_t)(((float)xv[j]     - mean) * rstd * w0[j] + b0[j]);
      u.e[hh * 8 + 4 + j] = (bf16_t)(((float)xv[4 + j] - mean) * rstd * w1[j] + b1[j]);
    }
  }
  return u.v;
}

// B fragment with fused LN (symmetric Gram: B row n == normalized row n).
__device__ __forceinline__ v16bf frag_ld_b_ln(const AS3 bf16_t* M, int ld, int n0, int k0,
                                              const AS3 float* st,
                                              const AS1 float* w, const AS1 float* b) {
  int l   = threadIdx.x & 31;
  int row = n0 + (l & 15);
  int ks  = k0 + ((l & 16) ? 8 : 0);
  float mean = st[2 * row], rstd = st[2 * row + 1];
  const AS3 bf16_t* p = M + (size_t)row * ld + ks;
  FragU u;
  v8bf x0 = ld8(p), x1 = ld8(p + 16);
#pragma unroll
  for (int q = 0; q < 4; ++q) {
    v4f wv = ld4f(w + ks + 4 * q + (q < 2 ? 0 : 8));
    v4f bv = ld4f(b + ks + 4 * q + (q < 2 ? 0 : 8));
#pragma unroll
    for (int j = 0; j < 4; ++j) {
      float x = (q < 2) ? (float)x0[4 * q + j] : (float)x1[4 * (q - 2) + j];
      u.e[4 * q + j] = (bf16_t)((x - mean) * rstd * wv[j] + bv[j]);
    }
  }
  return u.v;
}

// generic WMMA gemm: output tiles distributed across the 8 waves.
template <typename FA, typename FB, typename FS>
__device__ __forceinline__ void gemm16(int Mt, int Nt, int Kt, FA la, FB lb, FS st) {
  int wave = threadIdx.x >> 5;
  for (int t = wave; t < Mt * Nt; t += NWAVES) {
    int mi = t / Nt, ni = t - mi * Nt;
    v8f acc = {};
    for (int kk = 0; kk < Kt; ++kk) {
      v16bf a = la(mi * 16, kk * 32);
      v16bf b = lb(ni * 16, kk * 32);
      acc = __builtin_amdgcn_wmma_f32_16x16x32_bf16(false, a, false, b, (short)0, acc, false, false);
    }
    st(mi * 16, ni * 16, acc);
  }
}

// D-tile -> bf16 with per-column bias; only the edge tile pays the row guard.
__device__ __forceinline__ void store_bias_bf16(AS3 bf16_t* D, int ld, const AS1 float* bias,
                                                int m0, int n0, v8f acc) {
  int l   = threadIdx.x & 31;
  int col = n0 + (l & 15);
  int rb  = m0 + ((l & 16) ? 8 : 0);
  float bv = bias[col];
  if (m0 + 16 <= 100) {
#pragma unroll
    for (int r = 0; r < 8; ++r) D[(size_t)(rb + r) * ld + col] = (bf16_t)(acc[r] + bv);
  } else {
#pragma unroll
    for (int r = 0; r < 8; ++r) {
      int row = rb + r;
      if (row < 100) D[(size_t)row * ld + col] = (bf16_t)(acc[r] + bv);
    }
  }
}

// per-row mean / rstd over C channels; rows 100..111 get (0,1).
__device__ void row_stats(const AS3 bf16_t* M, int C, int ld, AS3 float* st) {
  int wave = threadIdx.x >> 5, lane = threadIdx.x & 31;
  for (int r = wave; r < 112; r += NWAVES) {
    float s = 0.f, s2 = 0.f;
    if (r < 100) {
      for (int c = lane; c < C; c += 32) {
        float x = (float)M[(size_t)r * ld + c];
        s += x; s2 += x * x;
      }
#pragma unroll
      for (int o = 16; o > 0; o >>= 1) {
        s  += __shfl_xor(s,  o, 32);
        s2 += __shfl_xor(s2, o, 32);
      }
    }
    if (lane == 0) {
      if (r < 100) {
        float m = s / C;
        st[2 * r]     = m;
        st[2 * r + 1] = rsqrtf(fmaxf(s2 / C - m * m, 0.f) + 1e-6f);
      } else {
        st[2 * r] = 0.f; st[2 * r + 1] = 1.f;
      }
    }
  }
}

// density: d[n] = sum_m cos(xn_n, xn_m), xn = LN(M). Gram via WMMA; norms
// from the Gram diagonal. Leaves row stats of M in `st`.
__device__ void density_calc(const AS3 bf16_t* M, int C, int ld,
                             const AS1 float* w, const AS1 float* b,
                             AS3 float* st, AS3 float* GR, AS3 float* dens, AS3 float* nrm) {
  row_stats(M, C, ld, st);
  __syncthreads();
  int Kt = pad32i(C) / 32;
  gemm16(7, 7, Kt,
    [&](int r0, int k0) { return frag_ld_a_ln(M, ld, r0, k0, st, w, b); },
    [&](int n0, int k0) { return frag_ld_b_ln(M, ld, n0, k0, st, w, b); },
    [&](int m0, int n0, v8f acc) {
      int l = threadIdx.x & 31;
      int col = n0 + (l & 15);
      int rb  = m0 + ((l & 16) ? 8 : 0);
      if (col < 100) {
        if (m0 + 16 <= 100) {
#pragma unroll
          for (int r = 0; r < 8; ++r) GR[(rb + r) * 112 + col] = acc[r];
        } else {
#pragma unroll
          for (int r = 0; r < 8; ++r) {
            int row = rb + r;
            if (row < 100) GR[row * 112 + col] = acc[r];
          }
        }
      }
    });
  __syncthreads();
  int tid = threadIdx.x;
  if (tid < 100) nrm[tid] = sqrtf(fmaxf(GR[tid * 112 + tid], 0.f));
  __syncthreads();
  if (tid < 100) {
    float ni = nrm[tid], s = 0.f;
    for (int m = 0; m < 100; ++m) s += GR[tid * 112 + m] / (ni * nrm[m] + 1e-8f);
    dens[tid] = s;
  }
  __syncthreads();
}

// ----------------------------- prep kernels ---------------------------------
// W[in][out] f32 -> Wt[out][pad32(in)] bf16 (zero padded along in).
__global__ void transpose_w(const float* __restrict__ W, bf16_t* __restrict__ Wt,
                            int In, int Out, int InPad) {
  int idx = blockIdx.x * 256 + threadIdx.x;
  if (idx >= Out * InPad) return;
  int o = idx / InPad, ii = idx - o * InPad;
  Wt[idx] = (ii < In) ? (bf16_t)W[(size_t)ii * Out + o] : (bf16_t)0.f;
}
// zero-padded f32 copy for LN params
__global__ void pad_f32(const float* __restrict__ src, float* __restrict__ dst,
                        int C, int CPad) {
  int i = blockIdx.x * 256 + threadIdx.x;
  if (i < CPad) dst[i] = (i < C) ? src[i] : 0.f;
}

// ------------------------------- main kernel --------------------------------
__global__ __launch_bounds__(NTHREADS) void model_main(KParams P) {
  __shared__ __align__(16) unsigned char smem[LDS_TOTAL];
  AS3 unsigned char* sb = (AS3 unsigned char*)smem;
  const int b   = blockIdx.x;
  const int tid = threadIdx.x;

  AS3 bf16_t* T0    = (AS3 bf16_t*)sb;
  AS3 bf16_t* TN    = (AS3 bf16_t*)(sb + TN_OFF);
  AS3 bf16_t* Hbuf  = (AS3 bf16_t*)(sb + H_OFF);
  AS3 float*  GR    = (AS3 float*)(sb + H_OFF);
  AS3 float*  CCRAW = (AS3 float*)(sb + CCRAW_OFF);
  AS3 bf16_t* Pb    = (AS3 bf16_t*)(sb + P_OFF);
  AS3 float*  cc    = (AS3 float*)(sb + SM_OFF);  // 448
  AS3 float*  cc2   = cc  + 448;                  // 448
  AS3 float*  dens  = cc2 + 448;                  // 112
  AS3 float*  nrm   = dens + 112;                 // 112
  AS3 float*  st0   = nrm  + 112;                 // 224
  AS3 float*  st1   = st0  + 224;                 // 224
  AS3 float*  sm    = st1  + 224;                 // 8 scalars

  const AS1 float* xb = (const AS1 float*)P.x + (size_t)b * 96 * 1400;
  const float alphaV = (float)P.alpha[0];

  {  // warm x through the cache hierarchy (global_prefetch_b8)
    const char* xc = (const char*)(P.x + (size_t)b * 96 * 1400);
#pragma unroll
    for (int it = 0; it < 8; ++it)
      __builtin_prefetch(xc + (size_t)(it * NTHREADS + tid) * 256, 0, 1);
  }

  // ---------------- embed: GEMM [1408 x 96] x [96 x 64] -> t0 [100 x 896]
  gemm16(88, 4, 3,
    [&](int r0, int k0) {
      int l  = tid & 31;
      int p  = r0 + (l & 15);
      int pc = (p < 1400) ? p : 0;      // branch-free: clamp + mask
      float msk = (p < 1400) ? 1.f : 0.f;
      int kb = k0 + ((l & 16) ? 8 : 0);
      FragU u;
#pragma unroll
      for (int hh = 0; hh < 2; ++hh)
#pragma unroll
        for (int j = 0; j < 8; ++j) {
          int c = kb + hh * 16 + j;
          u.e[hh * 8 + j] = (bf16_t)(xb[(size_t)c * 1400 + pc] * msk);
        }
      return u.v;
    },
    [&](int n0, int k0) { return frag_ld_b((const AS1 bf16_t*)P.wt_embed, 96, n0, k0); },
    [&](int m0, int n0, v8f acc) {
      int l  = tid & 31;
      int o  = n0 + (l & 15);
      int pb = m0 + ((l & 16) ? 8 : 0);
      // p -> (n,f) split: p = pb + r, rows advance within one n
      // except across multiples of 14; n = p / 14, f = p % 14 computed per r.
      float bias = ((const AS1 float*)P.embed_b)[o];
      if (m0 + 16 <= 1400) {   // 87 of 88 tiles: no bounds check at all
#pragma unroll
        for (int r = 0; r < 8; ++r) {
          int p = pb + r;
          int n = p / 14, f = p - n * 14;
          T0[(size_t)n * 896 + o * 14 + f] = (bf16_t)(acc[r] + bias);
        }
      } else {
#pragma unroll
        for (int r = 0; r < 8; ++r) {
          int p = pb + r;
          if (p < 1400) {
            int n = p / 14, f = p - n * 14;
            T0[(size_t)n * 896 + o * 14 + f] = (bf16_t)(acc[r] + bias);
          }
        }
      }
    });
  __syncthreads();

  // ---------------- ccg: density on LN(t0), argmax mask sum, proj+relu
  density_calc(T0, 896, 896, (const AS1 float*)P.ccg_lnw, (const AS1 float*)P.ccg_lnb,
               st0, GR, dens, nrm);
  if (tid == 0) {
    float mx = -1e30f;
    for (int n = 0; n < 100; ++n) mx = fmaxf(mx, dens[n]);
    sm[0] = mx;
  }
  __syncthreads();
  {
    const AS1 float* lw = (const AS1 float*)P.ccg_lnw;
    const AS1 float* lb = (const AS1 float*)P.ccg_lnb;
    float dmax = sm[0];
    for (int c = tid; c < 896; c += NTHREADS) {
      float wv = lw[c], bv = lb[c], acc = 0.f;
      for (int n = 0; n < 100; ++n) {
        float msk = (dens[n] == dmax) ? 1.f : 0.f;
        acc += msk * (((float)T0[(size_t)n * 896 + c] - st0[2 * n]) * st0[2 * n + 1] * wv + bv);
      }
      CCRAW[c] = acc;
    }
  }
  __syncthreads();
  {
    const AS1 float* pb = (const AS1 float*)P.ccg_pb;
    for (int o = tid; o < 448; o += NTHREADS) {
      float acc = pb[o];
      const AS1 bf16_t* wr = (const AS1 bf16_t*)P.wt_ccg + (size_t)o * 896;
      for (int c = 0; c < 896; ++c) acc += CCRAW[c] * (float)wr[c];
      cc[o] = fmaxf(acc, 0.f);
    }
  }
  __syncthreads();

  // ---------------- 4 blocks + ccs ------------------------------------------
  const int dArr[4]  = {896, 448, 224, 112};
  const int c2Arr[4] = {448, 224, 112, 56};

  for (int i = 0; i < 4; ++i) {
    const int d   = dArr[i], c2 = c2Arr[i];
    const int ldt = pad32i(d);
    const int ldh = pad32i(c2);
    AS3 bf16_t* Tin = (i == 0) ? T0 : TN;
    AS3 bf16_t* Qb  = (AS3 bf16_t*)sb;
    AS3 bf16_t* Kb  = (AS3 bf16_t*)(sb + (size_t)100 * ldh * 2);
    AS3 bf16_t* VT  = (AS3 bf16_t*)sb;  // [c2][128], over dead Q/K

    // h = LN(t) @ red_w + red_b   (st0 = t row stats from preceding density)
    gemm16(7, c2 / 16, ldt / 32,
      [&](int r0, int k0) {
        return frag_ld_a_ln(Tin, ldt, r0, k0, st0,
                            (const AS1 float*)P.n0w[i], (const AS1 float*)P.n0b[i]);
      },
      [&](int n0, int k0) { return frag_ld_b((const AS1 bf16_t*)P.wt_red[i], ldt, n0, k0); },
      [&](int m0, int n0, v8f acc) {
        store_bias_bf16(Hbuf, ldh, (const AS1 float*)P.redb[i], m0, n0, acc);
      });
    __syncthreads();

    // zero pad columns of h (stale f32 Gram bytes there could be bf16-NaN)
    int padc = ldh - c2;
    if (padc > 0) {
      for (int idx = tid; idx < 100 * padc; idx += NTHREADS) {
        int r = idx / padc, c = c2 + (idx - r * padc);
        Hbuf[(size_t)r * ldh + c] = (bf16_t)0.f;
      }
    }
    __syncthreads();

    row_stats(Hbuf, c2, ldh, st1);
    __syncthreads();

    auto lnH = [&](int r0, int k0) {
      return frag_ld_a_ln(Hbuf, ldh, r0, k0, st1,
                          (const AS1 float*)P.n1w[i], (const AS1 float*)P.n1b[i]);
    };

    // q, k
    gemm16(7, c2 / 16, ldh / 32, lnH,
      [&](int n0, int k0) { return frag_ld_b((const AS1 bf16_t*)P.wt_q[i], ldh, n0, k0); },
      [&](int m0, int n0, v8f acc) {
        store_bias_bf16(Qb, ldh, (const AS1 float*)P.qb[i], m0, n0, acc);
      });
    gemm16(7, c2 / 16, ldh / 32, lnH,
      [&](int n0, int k0) { return frag_ld_b((const AS1 bf16_t*)P.wt_k[i], ldh, n0, k0); },
      [&](int m0, int n0, v8f acc) {
        store_bias_bf16(Kb, ldh, (const AS1 float*)P.kb[i], m0, n0, acc);
      });
    __syncthreads();
    if (padc > 0) {  // zero K-dim padding of q/k so padded-K score GEMM is exact
      for (int idx = tid; idx < 100 * padc; idx += NTHREADS) {
        int r = idx / padc, c = c2 + (idx - r * padc);
        Qb[(size_t)r * ldh + c] = (bf16_t)0.f;
        Kb[(size_t)r * ldh + c] = (bf16_t)0.f;
      }
      __syncthreads();
    }

    // scores = q k^T * c2^-0.5  (B fragments read k row-major directly)
    float scale = rsqrtf((float)c2);
    gemm16(7, 7, ldh / 32,
      [&](int r0, int k0) { return frag_ld_a(Qb, ldh, r0, k0); },
      [&](int n0, int k0) { return frag_ld_b(Kb, ldh, n0, k0); },
      [&](int m0, int n0, v8f acc) {
        int l = tid & 31;
        int col = n0 + (l & 15);
        int rb  = m0 + ((l & 16) ? 8 : 0);
        if (col < 100) {
          if (m0 + 16 <= 100) {
#pragma unroll
            for (int r = 0; r < 8; ++r) Pb[(rb + r) * 128 + col] = (bf16_t)(acc[r] * scale);
          } else {
#pragma unroll
            for (int r = 0; r < 8; ++r) {
              int row = rb + r;
              if (row < 100) Pb[row * 128 + col] = (bf16_t)(acc[r] * scale);
            }
          }
        }
      });
    __syncthreads();

    // softmax rows; zero K-padding cols 100..127
    if (tid < 100) {
      AS3 bf16_t* rowp = Pb + tid * 128;
      float mx = -1e30f;
      for (int c = 0; c < 100; ++c) mx = fmaxf(mx, (float)rowp[c]);
      float s = 0.f;
      for (int c = 0; c < 100; ++c) s += __expf((float)rowp[c] - mx);
      float inv = 1.f / s;
      for (int c = 0; c < 100; ++c) rowp[c] = (bf16_t)(__expf((float)rowp[c] - mx) * inv);
      for (int c = 100; c < 128; ++c) rowp[c] = (bf16_t)0.f;
    }
    __syncthreads();

    // v, stored transposed: VT[c][m]
    gemm16(7, c2 / 16, ldh / 32, lnH,
      [&](int n0, int k0) { return frag_ld_b((const AS1 bf16_t*)P.wt_v[i], ldh, n0, k0); },
      [&](int m0, int n0, v8f acc) {
        int l = tid & 31;
        int col = n0 + (l & 15);
        int rb  = m0 + ((l & 16) ? 8 : 0);
        float bv = ((const AS1 float*)P.vb[i])[col];
        if (m0 + 16 <= 100) {
#pragma unroll
          for (int r = 0; r < 8; ++r) VT[(size_t)col * 128 + rb + r] = (bf16_t)(acc[r] + bv);
        } else {
#pragma unroll
          for (int r = 0; r < 8; ++r) {
            int row = rb + r;
            if (row < 100) VT[(size_t)col * 128 + row] = (bf16_t)(acc[r] + bv);
          }
        }
      });
    __syncthreads();
    for (int idx = tid; idx < c2 * 28; idx += NTHREADS) {  // zero VT[:,100..127]
      int r = idx / 28, m = 100 + (idx - r * 28);
      VT[(size_t)r * 128 + m] = (bf16_t)0.f;
    }
    __syncthreads();

    // new t = h + probs @ v
    gemm16(7, c2 / 16, 4,
      [&](int r0, int k0) { return frag_ld_a(Pb, 128, r0, k0); },
      [&](int n0, int k0) { return frag_ld_b(VT, 128, n0, k0); },
      [&](int m0, int n0, v8f acc) {
        int l = tid & 31;
        int col = n0 + (l & 15);
        int rb  = m0 + ((l & 16) ? 8 : 0);
        if (m0 + 16 <= 100) {
#pragma unroll
          for (int r = 0; r < 8; ++r) {
            size_t ix = (size_t)(rb + r) * ldh + col;
            TN[ix] = (bf16_t)(acc[r] + (float)Hbuf[ix]);
          }
        } else {
#pragma unroll
          for (int r = 0; r < 8; ++r) {
            int row = rb + r;
            if (row < 100) {
              size_t ix = (size_t)row * ldh + col;
              TN[ix] = (bf16_t)(acc[r] + (float)Hbuf[ix]);
            }
          }
        }
      });
    __syncthreads();

    // ---------------- ccs(i) -------------------------------------------------
    density_calc(TN, c2, ldh, (const AS1 float*)P.cs_lnw[i], (const AS1 float*)P.cs_lnb[i],
                 st0, GR, dens, nrm);
    if (tid == 0) {
      float mn = 1e30f, mx = -1e30f;
      for (int n = 0; n < 100; ++n) { mn = fminf(mn, dens[n]); mx = fmaxf(mx, dens[n]); }
      sm[0] = mn; sm[1] = mx;
    }
    __syncthreads();
    if (tid < 100) dens[tid] = (dens[tid] - sm[0]) / (sm[1] - sm[0] + 1e-8f);
    __syncthreads();
    if (tid == 0) {
      const AS1 float* tw = (const AS1 float*)P.thw[i];
      float a = P.thb[i][0];
      for (int n = 0; n < 100; ++n) a += dens[n] * tw[n];
      sm[2] = (1.f / (1.f + __expf(-a))) * alphaV;
    }
    __syncthreads();
    if (tid < 100) dens[tid] = fmaxf(dens[tid] - sm[2], 0.f);
    __syncthreads();
    if (tid == 0) {
      float s = 0.f;
      for (int n = 0; n < 100; ++n) s += dens[n];
      sm[3] = s + 1e-8f;
      sm[4] = s / (s + 1e-8f);  // sum of dt
    }
    __syncthreads();
    if (tid < 100) dens[tid] /= sm[3];
    __syncthreads();
    {
      const AS1 float* lw = (const AS1 float*)P.cs_lnw[i];
      const AS1 float* lb = (const AS1 float*)P.cs_lnb[i];
      float sumdt = sm[4];
      for (int c = tid; c < c2; c += NTHREADS) {
        float wv = lw[c], bv = lb[c], acc = 0.f;
        for (int n = 0; n < 100; ++n) {
          float xn = ((float)TN[(size_t)n * ldh + c] - st0[2 * n]) * st0[2 * n + 1] * wv + bv;
          acc += xn * dens[n];
        }
        cc2[c] = cc[c] + (acc - cc[c] * sumdt) * 0.01f;
      }
    }
    __syncthreads();
    if (i < 3) {
      int half = c2 / 2, ldw = pad32i(c2);
      const AS1 float* pb = (const AS1 float*)P.cs_pb[i];
      for (int o = tid; o < half; o += NTHREADS) {
        float acc = pb[o];
        const AS1 bf16_t* wr = (const AS1 bf16_t*)P.wt_cp[i] + (size_t)o * ldw;
        for (int c = 0; c < c2; ++c) acc += cc2[c] * (float)wr[c];
        cc[o] = fmaxf(acc, 0.f);
      }
    } else if (tid < 56) {
      cc[tid] = cc2[tid];
    }
    __syncthreads();
  }

  // ---------------- head -----------------------------------------------------
  if (tid == 0) {
    float m = 0.f;
    for (int j = 0; j < 56; ++j) m += cc[j];
    m /= 56.f;
    float v = 0.f;
    for (int j = 0; j < 56; ++j) { float dv = cc[j] - m; v += dv * dv; }
    float rs = rsqrtf(v / 56.f + 1e-6f);
    float acc = P.headb[0];
    for (int j = 0; j < 56; ++j)
      acc += ((cc[j] - m) * rs * P.normw[j] + P.normb[j]) * P.headw[j];
    P.out[b] = 1.f / (1.f + __expf(-acc));
  }
}

// ------------------------------- host launcher ------------------------------
extern "C" void kernel_launch(void* const* d_in, const int* in_sizes, int n_in,
                              void* d_out, int out_size, void* d_ws, size_t ws_size,
                              hipStream_t stream) {
  (void)in_sizes; (void)n_in; (void)out_size; (void)ws_size;
  auto F = [&](int i) { return (const float*)d_in[i]; };

  // setup_inputs() insertion-order flattening:
  // 0:x 1:alpha 2:embed_w 3:embed_b 4..7:ccg{ln_w,ln_b,proj_w,proj_b}
  // 8+12i: block i {norm0_w,norm0_b,red_w,red_b,norm1_w,norm1_b,q_w,q_b,k_w,k_b,v_w,v_b}
  // 56..: ccs i {ln_w,ln_b,th_w,th_b[,proj_w,proj_b]}  78..81: norm/head
  KParams P{};
  P.x = F(0); P.alpha = (const int*)d_in[1]; P.out = (float*)d_out;
  P.embed_b = F(3);
  P.ccg_pb = F(7);
  const int dA[4] = {896, 448, 224, 112};
  for (int i = 0; i < 4; ++i) {
    int base = 8 + 12 * i;
    P.redb[i] = F(base + 3);
    P.qb[i] = F(base + 7); P.kb[i] = F(base + 9); P.vb[i] = F(base + 11);
  }
  {
    int cbase = 56;
    for (int i = 0; i < 4; ++i) {
      P.thw[i] = F(cbase + 2); P.thb[i] = F(cbase + 3);
      if (i < 3) { P.cs_pb[i] = F(cbase + 5); cbase += 6; } else cbase += 4;
    }
  }
  P.normw = F(78); P.normb = F(79); P.headw = F(80); P.headb = F(81);

  // --- prep: transposed bf16 weights (K zero-padded to 32) ---
  bf16_t* ws = (bf16_t*)d_ws;
  size_t  off = 0;
  auto addW = [&](const float* W, int In, int Out) -> const bf16_t* {
    const bf16_t* dst = ws + off;
    int    inPad = pad32i(In);
    size_t total = (size_t)Out * inPad;
    int    grid  = (int)((total + 255) / 256);
    transpose_w<<<grid, 256, 0, stream>>>(W, ws + off, In, Out, inPad);
    off += total;
    return dst;
  };
  P.wt_embed = addW(F(2), 96, 64);
  P.wt_ccg   = addW(F(6), 896, 448);
  for (int i = 0; i < 4; ++i) {
    int d = dA[i], c2 = d / 2, base = 8 + 12 * i;
    P.wt_red[i] = addW(F(base + 2), d,  c2);
    P.wt_q[i]   = addW(F(base + 6), c2, c2);
    P.wt_k[i]   = addW(F(base + 8), c2, c2);
    P.wt_v[i]   = addW(F(base + 10), c2, c2);
  }
  {
    const int pidx[3] = {60, 66, 72};
    const int c2s[3]  = {448, 224, 112};
    for (int i = 0; i < 3; ++i) P.wt_cp[i] = addW(F(pidx[i]), c2s[i], c2s[i] / 2);
  }

  // --- prep: zero-padded f32 LN params (16B-aligned region after bf16 part) ---
  size_t fbyte = (2 * off + 15) & ~(size_t)15;
  float* wsf = (float*)((char*)d_ws + fbyte);
  size_t foff = 0;
  auto addP = [&](const float* src, int C) -> const float* {
    const float* dst = wsf + foff;
    int CPad = pad32i(C);
    pad_f32<<<(CPad + 255) / 256, 256, 0, stream>>>(src, wsf + foff, C, CPad);
    foff += CPad;
    return dst;
  };
  P.ccg_lnw = addP(F(4), 896);
  P.ccg_lnb = addP(F(5), 896);
  for (int i = 0; i < 4; ++i) {
    int base = 8 + 12 * i, d = dA[i], c2 = d / 2;
    P.n0w[i] = addP(F(base + 0), d);
    P.n0b[i] = addP(F(base + 1), d);
    P.n1w[i] = addP(F(base + 4), c2);
    P.n1b[i] = addP(F(base + 5), c2);
  }
  {
    int cbase = 56;
    const int c2s[4] = {448, 224, 112, 56};
    for (int i = 0; i < 4; ++i) {
      P.cs_lnw[i] = addP(F(cbase), c2s[i]);
      P.cs_lnb[i] = addP(F(cbase + 1), c2s[i]);
      cbase += (i < 3) ? 6 : 4;
    }
  }

  model_main<<<dim3(1024), dim3(NTHREADS), 0, stream>>>(P);
}
